// Model_38328288149535
// MI455X (gfx1250) — matrix operand order repaired and checked by reading, not compiled, not run on hardware
//
#include <hip/hip_runtime.h>
#include <hip/hip_fp16.h>

// MI455X / gfx1250, wave32. WMMA f16 16x16x32 for all LSTM GEMMs.

typedef __attribute__((ext_vector_type(16))) _Float16 v16h;
typedef __attribute__((ext_vector_type(8)))  _Float16 v8h;
typedef __attribute__((ext_vector_type(8)))  float    v8f;

#define HDIM 256
#define GDIM 1024   // 4 gates * 256

__device__ __forceinline__ float sigmoidf_(float x) { return 1.0f / (1.0f + __expf(-x)); }

// ---------------------------------------------------------------------------
// Gather word embeddings (and title embeddings) into fp16, K padded 300->320.
// One block per (sequence,timestep) row, 320 threads.
// ---------------------------------------------------------------------------
__global__ __launch_bounds__(320) void gather_embed(
    const int* __restrict__ ids, const int* __restrict__ titles,
    const float* __restrict__ emb,
    _Float16* __restrict__ words, _Float16* __restrict__ twords)
{
  int r = blockIdx.x;
  int k = threadIdx.x;                  // 0..319
  if (r < 640 * 30) {
    int id = ids[r];
    words[(size_t)r * 320 + k] = (k < 300) ? (_Float16)emb[(size_t)id * 300 + k] : (_Float16)0.f;
  } else {
    int r2 = r - 640 * 30;
    int id = titles[r2];
    twords[(size_t)r2 * 320 + k] = (k < 300) ? (_Float16)emb[(size_t)id * 300 + k] : (_Float16)0.f;
  }
}

// ---------------------------------------------------------------------------
// Re-tile an LSTM weight matrix src[N][Kreal] (f32, N=4H gate-major rows) into
// a WMMA B-fragment layout (f16, contiguous k per lane half, same order as the A loads). One 1KB tile per (32K x 16N):
//   tile = kt*(N/16)+nt ; lane l holds n = nt*16+(l&15), k = kt*32+(l>>4)*16+i
// so a lane's whole fragment is one contiguous 32B load at tile*512 + lane*16.
// 128 threads = 4 tiles per block.
// ---------------------------------------------------------------------------
__global__ __launch_bounds__(128) void tile_weight(
    const float* __restrict__ src, _Float16* __restrict__ dst,
    int Kreal, int Kpad, int N)
{
  int lane = threadIdx.x & 31;
  int tile = blockIdx.x * 4 + (threadIdx.x >> 5);
  int ntiles = N >> 4;
  int kt = tile / ntiles, nt = tile % ntiles;
  int n = nt * 16 + (lane & 15);
  int kbase = kt * 32 + (lane >> 4) * 16;
  _Float16* d = dst + (size_t)tile * 512 + lane * 16;
#pragma unroll
  for (int i = 0; i < 16; ++i) {
    int k = kbase + i;
    float v = (k < Kreal) ? src[(size_t)n * Kreal + k] : 0.f;
    d[i] = (_Float16)v;
  }
  (void)Kpad;
}

// ---------------------------------------------------------------------------
// Generic single-layer LSTM direction-pair kernel (wave32 WMMA).
// grid.x = nseq/16 chunks, grid.y = direction (0 fwd, 1 bwd).
// Block = 128 threads = 4 waves; wave w owns gate w (i,f,g,o), i.e. output
// columns [w*256, w*256+256) split into 16 N-tiles of 16.
// Per step: gates[16,1024] = x_t[16,Kx] @ WihT + h[16,256] @ WhhT + b
// via v_wmma_f32_16x16x32_f16, then activation -> LDS, then c/h update.
// x is fp16 [nseq][T][Kx] (Kx multiple of 32); h kept fp16 in LDS for A-frags.
// ---------------------------------------------------------------------------
__global__ __launch_bounds__(128) void lstm_kernel(
    const _Float16* __restrict__ x, int T, int Kx,
    const _Float16* __restrict__ wihT_f, const _Float16* __restrict__ whhT_f, const float* __restrict__ bias_f,
    const _Float16* __restrict__ wihT_b, const _Float16* __restrict__ whhT_b, const float* __restrict__ bias_b,
    float* __restrict__ out_all,      // [nseq][T][512] (+dir*256), time-reversed for bwd; may be null
    float* __restrict__ out_final)    // [nseq][512] (+dir*256); may be null
{
  __shared__ _Float16 hbuf[16 * HDIM];   // 8 KB, current hidden (A-matrix source)
  __shared__ _Float16 gbuf[16 * GDIM];   // 32 KB, activated gates

  const int tid  = threadIdx.x;
  const int lane = tid & 31;
  const int wv   = tid >> 5;            // wave id == gate id
  const int dir  = blockIdx.y;
  const int seq0 = blockIdx.x * 16;

  const _Float16* wihT = dir ? wihT_b : wihT_f;
  const _Float16* whhT = dir ? whhT_b : whhT_f;
  const float*    bias = dir ? bias_b  : bias_f;

  const int m_a = lane & 15;            // A-matrix row / C-matrix column owner
  const int kh  = lane >> 4;            // K-half select for A/B fragments
  const int KI  = Kx >> 5;

  for (int i = tid; i < 16 * HDIM; i += 128) hbuf[i] = (_Float16)0.f;
  float c_reg[32];
#pragma unroll
  for (int i = 0; i < 32; ++i) c_reg[i] = 0.f;
  const int um = tid & 15, uj0 = (tid >> 4) * 32;   // update-phase ownership
  __syncthreads();

  for (int t = 0; t < T; ++t) {
    const int tx = dir ? (T - 1 - t) : t;
    const _Float16* xrow = x + ((size_t)(seq0 + m_a) * T + tx) * Kx;

    for (int ntl = 0; ntl < 16; ++ntl) {
      const int n0 = wv * 256 + ntl * 16;
      const float bv = bias[n0 + m_a];
      v8f acc;
#pragma unroll
      for (int r = 0; r < 8; ++r) acc[r] = bv;

      // input projection: x_t[16,Kx] @ WihT -> A frags from global fp16 words
      for (int kk = 0; kk < KI; ++kk) {
        v8h alo = *(const v8h*)(xrow + kk * 32 + kh * 16);
        v8h ahi = *(const v8h*)(xrow + kk * 32 + kh * 16 + 8);
        v16h a;
#pragma unroll
        for (int i = 0; i < 8; ++i) { a[i] = alo[i]; a[i + 8] = ahi[i]; }
        v16h bf = *(const v16h*)(wihT + (size_t)(kk * 64 + (n0 >> 4)) * 512 + lane * 16);
        acc = __builtin_amdgcn_wmma_f32_16x16x32_f16(false, a, false, bf, (short)0, acc, false, false);
      }
      // recurrent part: h[16,256] @ WhhT -> A frags from LDS
#pragma unroll
      for (int kk = 0; kk < 8; ++kk) {
        v8h alo = *(const v8h*)&hbuf[m_a * HDIM + kk * 32 + kh * 16];
        v8h ahi = *(const v8h*)&hbuf[m_a * HDIM + kk * 32 + kh * 16 + 8];
        v16h a;
#pragma unroll
        for (int i = 0; i < 8; ++i) { a[i] = alo[i]; a[i + 8] = ahi[i]; }
        v16h bf = *(const v16h*)(whhT + (size_t)(kk * 64 + (n0 >> 4)) * 512 + lane * 16);
        acc = __builtin_amdgcn_wmma_f32_16x16x32_f16(false, a, false, bf, (short)0, acc, false, false);
      }
      // activate (wave 2 == g-gate -> tanh, else sigmoid) and spill to LDS.
      // C/D layout: VGPR r holds M = r + (lane>=16 ? 8 : 0), N = n0 + (lane&15).
#pragma unroll
      for (int r = 0; r < 8; ++r) {
        int mm = r + kh * 8;
        float v = acc[r];
        float av = (wv == 2) ? tanhf(v) : sigmoidf_(v);
        gbuf[mm * GDIM + n0 + m_a] = (_Float16)av;
      }
    }
    __syncthreads();

    // cell/hidden update: thread owns (m=um, j in [uj0, uj0+32))
    for (int jj = 0; jj < 32; ++jj) {
      int j = uj0 + jj;
      float ig = (float)gbuf[um * GDIM + j];
      float fg = (float)gbuf[um * GDIM + 256 + j];
      float gg = (float)gbuf[um * GDIM + 512 + j];
      float og = (float)gbuf[um * GDIM + 768 + j];
      float c = fg * c_reg[jj] + ig * gg;
      c_reg[jj] = c;
      float h = og * tanhf(c);
      hbuf[um * HDIM + j] = (_Float16)h;
      if (out_all) {
        int tout = dir ? (T - 1 - t) : t;
        out_all[((size_t)(seq0 + um) * T + tout) * 512 + dir * 256 + j] = h;
      }
      if (out_final && t == T - 1)
        out_final[(size_t)(seq0 + um) * 512 + dir * 256 + j] = h;
    }
    __syncthreads();
  }
}

// ---------------------------------------------------------------------------
// SourceBias: out[t] = tanh(sen[t] @ trans[u_t] + sb_bias[u_t]).
// Each token has its own 512x512 matrix -> batched matvec, coalesced over e.
// ---------------------------------------------------------------------------
__global__ __launch_bounds__(256) void sourcebias_kernel(
    const float* __restrict__ sentences, const int* __restrict__ urls,
    const float* __restrict__ trans, const float* __restrict__ sb_bias,
    _Float16* __restrict__ biased)
{
  __shared__ float sv[512];
  int t = blockIdx.x, tid = threadIdx.x;
  sv[tid]       = sentences[(size_t)t * 512 + tid];
  sv[tid + 256] = sentences[(size_t)t * 512 + 256 + tid];
  __syncthreads();
  int u = urls[t];
  const float* tr = trans + (size_t)u * 512 * 512;
  for (int e = tid; e < 512; e += 256) {
    float acc = sb_bias[(size_t)u * 512 + e];
    for (int d = 0; d < 512; ++d) acc += sv[d] * tr[(size_t)d * 512 + e];
    biased[(size_t)t * 512 + e] = (_Float16)tanhf(acc);
  }
}

// ---------------------------------------------------------------------------
// Bilinear attention + both MLP heads. One block per document (b).
// ---------------------------------------------------------------------------
__global__ __launch_bounds__(256) void attn_heads_kernel(
    const float* __restrict__ documents, const float* __restrict__ headings,
    const float* __restrict__ attn_W, const float* __restrict__ attn_b,
    const float* __restrict__ bW1, const float* __restrict__ bb1,
    const float* __restrict__ bW2, const float* __restrict__ bb2,
    const float* __restrict__ tW1, const float* __restrict__ tb1,
    const float* __restrict__ tW2, const float* __restrict__ tb2,
    float* __restrict__ out)
{
  __shared__ float head[512], tmp[512], rep[512], h1[256], t1s[256], sc[16], wgt[16], zz[8];
  int b = blockIdx.x, tid = threadIdx.x;
  head[tid]       = headings[(size_t)b * 512 + tid];
  head[tid + 256] = headings[(size_t)b * 512 + 256 + tid];
  __syncthreads();
  for (int d = tid; d < 512; d += 256) {              // tmp = W @ head
    float a = 0.f;
    for (int e = 0; e < 512; ++e) a += attn_W[(size_t)d * 512 + e] * head[e];
    tmp[d] = a;
  }
  __syncthreads();
  if (tid < 10) {                                     // scores
    float a = attn_b[0];
    const float* dd = documents + ((size_t)b * 10 + tid) * 512;
    for (int d = 0; d < 512; ++d) a += dd[d] * tmp[d];
    sc[tid] = a;
  }
  __syncthreads();
  if (tid == 0) {                                     // softmax over 10
    float mx = sc[0];
    for (int s = 1; s < 10; ++s) mx = fmaxf(mx, sc[s]);
    float sum = 0.f;
    for (int s = 0; s < 10; ++s) { float e = __expf(sc[s] - mx); wgt[s] = e; sum += e; }
    for (int s = 0; s < 10; ++s) wgt[s] /= sum;
  }
  __syncthreads();
  for (int d = tid; d < 512; d += 256) {              // doc_rep
    float a = 0.f;
    for (int s = 0; s < 10; ++s) a += documents[((size_t)b * 10 + s) * 512 + d] * wgt[s];
    rep[d] = a;
  }
  __syncthreads();
  {                                                   // both first-layer MLPs
    float ab = bb1[tid], at = tb1[tid];
    for (int d = 0; d < 512; ++d) {
      float rv = rep[d];
      ab += rv * bW1[(size_t)tid * 512 + d];
      at += rv * tW1[(size_t)tid * 512 + d];
    }
    h1[tid]  = tanhf(ab);
    t1s[tid] = tanhf(at);
  }
  __syncthreads();
  if (tid < 5) {
    float a = bb2[tid];
    for (int j = 0; j < 256; ++j) a += h1[j] * bW2[(size_t)tid * 256 + j];
    zz[tid] = tanhf(a);
  }
  if (tid == 5) {
    float a = tb2[0];
    for (int j = 0; j < 256; ++j) a += t1s[j] * tW2[j];
    out[64 * 5 + b] = sigmoidf_(tanhf(a));            // truth_out
  }
  __syncthreads();
  if (tid == 0) {                                     // softmax over 5 classes
    float mx = zz[0];
    for (int k = 1; k < 5; ++k) mx = fmaxf(mx, zz[k]);
    float sum = 0.f; float e[5];
    for (int k = 0; k < 5; ++k) { e[k] = __expf(zz[k] - mx); sum += e[k]; }
    for (int k = 0; k < 5; ++k) out[(size_t)b * 5 + k] = e[k] / sum;
  }
}

// ---------------------------------------------------------------------------
extern "C" void kernel_launch(void* const* d_in, const int* in_sizes, int n_in,
                              void* d_out, int out_size, void* d_ws, size_t ws_size,
                              hipStream_t stream) {
  const int*   input_ids = (const int*)d_in[0];
  const int*   urls      = (const int*)d_in[1];
  const int*   titles    = (const int*)d_in[2];
  const float* emb       = (const float*)d_in[3];
  const float* sen_wih_f = (const float*)d_in[4];
  const float* sen_whh_f = (const float*)d_in[5];
  const float* sen_b_f   = (const float*)d_in[6];
  const float* sen_wih_b = (const float*)d_in[7];
  const float* sen_whh_b = (const float*)d_in[8];
  const float* sen_b_b   = (const float*)d_in[9];
  const float* trans     = (const float*)d_in[10];
  const float* sb_bias   = (const float*)d_in[11];
  const float* doc_wih_f = (const float*)d_in[12];
  const float* doc_whh_f = (const float*)d_in[13];
  const float* doc_b_f   = (const float*)d_in[14];
  const float* doc_wih_b = (const float*)d_in[15];
  const float* doc_whh_b = (const float*)d_in[16];
  const float* doc_b_b   = (const float*)d_in[17];
  const float* attn_W    = (const float*)d_in[18];
  const float* attn_b    = (const float*)d_in[19];
  const float* bias_W1   = (const float*)d_in[20];
  const float* bias_b1   = (const float*)d_in[21];
  const float* bias_W2   = (const float*)d_in[22];
  const float* bias_b2   = (const float*)d_in[23];
  const float* truth_W1  = (const float*)d_in[24];
  const float* truth_b1  = (const float*)d_in[25];
  const float* truth_W2  = (const float*)d_in[26];
  const float* truth_b2  = (const float*)d_in[27];
  (void)in_sizes; (void)n_in; (void)out_size; (void)ws_size;

  // ---- workspace layout (all 256B aligned; total ~22 MB) ----
  char* ws = (char*)d_ws;
  size_t off = 0;
  auto alloc = [&](size_t bytes) -> char* {
    char* p = ws + off;
    off += (bytes + 255) & ~(size_t)255;
    return p;
  };
  _Float16* wsWords  = (_Float16*)alloc((size_t)640 * 30 * 320 * 2);  // gathered word embeds
  _Float16* wsTitles = (_Float16*)alloc((size_t)64 * 15 * 320 * 2);
  _Float16* sWihF = (_Float16*)alloc((size_t)10 * 64 * 512 * 2);      // tiled weights
  _Float16* sWihB = (_Float16*)alloc((size_t)10 * 64 * 512 * 2);
  _Float16* sWhhF = (_Float16*)alloc((size_t)8 * 64 * 512 * 2);
  _Float16* sWhhB = (_Float16*)alloc((size_t)8 * 64 * 512 * 2);
  _Float16* dWihF = (_Float16*)alloc((size_t)16 * 64 * 512 * 2);
  _Float16* dWihB = (_Float16*)alloc((size_t)16 * 64 * 512 * 2);
  _Float16* dWhhF = (_Float16*)alloc((size_t)8 * 64 * 512 * 2);
  _Float16* dWhhB = (_Float16*)alloc((size_t)8 * 64 * 512 * 2);
  float*    wsSent   = (float*)alloc((size_t)640 * 512 * 4);          // sentence final hidden
  float*    wsHead   = (float*)alloc((size_t)64 * 512 * 4);           // title final hidden
  _Float16* wsBiased = (_Float16*)alloc((size_t)640 * 512 * 2);       // SourceBias output (fp16)
  float*    wsDocs   = (float*)alloc((size_t)64 * 10 * 512 * 4);      // doc LSTM all outputs

  // 1) gather embeddings to fp16 (words + titles), K padded to 320
  gather_embed<<<640 * 30 + 64 * 15, 320, 0, stream>>>(input_ids, titles, emb, wsWords, wsTitles);

  // 2) tile all LSTM weights into WMMA B-fragment layout
  tile_weight<<<(10 * 64) / 4, 128, 0, stream>>>(sen_wih_f, sWihF, 300, 320, 1024);
  tile_weight<<<(10 * 64) / 4, 128, 0, stream>>>(sen_wih_b, sWihB, 300, 320, 1024);
  tile_weight<<<(8 * 64) / 4, 128, 0, stream>>>(sen_whh_f, sWhhF, 256, 256, 1024);
  tile_weight<<<(8 * 64) / 4, 128, 0, stream>>>(sen_whh_b, sWhhB, 256, 256, 1024);
  tile_weight<<<(16 * 64) / 4, 128, 0, stream>>>(doc_wih_f, dWihF, 512, 512, 1024);
  tile_weight<<<(16 * 64) / 4, 128, 0, stream>>>(doc_wih_b, dWihB, 512, 512, 1024);
  tile_weight<<<(8 * 64) / 4, 128, 0, stream>>>(doc_whh_f, dWhhF, 256, 256, 1024);
  tile_weight<<<(8 * 64) / 4, 128, 0, stream>>>(doc_whh_b, dWhhB, 256, 256, 1024);

  // 3) sentence BiLSTM: 640 seqs -> final hidden [640][512]
  lstm_kernel<<<dim3(40, 2), 128, 0, stream>>>(wsWords, 30, 320,
      sWihF, sWhhF, sen_b_f, sWihB, sWhhB, sen_b_b, nullptr, wsSent);

  // 4) title BiLSTM (same weights): 64 seqs of 15 -> headings [64][512]
  lstm_kernel<<<dim3(4, 2), 128, 0, stream>>>(wsTitles, 15, 320,
      sWihF, sWhhF, sen_b_f, sWihB, sWhhB, sen_b_b, nullptr, wsHead);

  // 5) per-url SourceBias
  sourcebias_kernel<<<640, 256, 0, stream>>>(wsSent, urls, trans, sb_bias, wsBiased);

  // 6) document BiLSTM: all timesteps -> documents [64][10][512]
  lstm_kernel<<<dim3(4, 2), 128, 0, stream>>>(wsBiased, 10, 512,
      dWihF, dWhhF, doc_b_f, dWihB, dWhhB, doc_b_b, wsDocs, nullptr);

  // 7) attention + heads -> d_out = [bias_out 64*5 | truth_out 64]
  attn_heads_kernel<<<64, 256, 0, stream>>>(wsDocs, wsHead, attn_W, attn_b,
      bias_W1, bias_b1, bias_W2, bias_b2, truth_W1, truth_b1, truth_W2, truth_b2,
      (float*)d_out);
}
